// CLIP_MLP_1262720385749
// MI455X (gfx1250) — matrix operand order repaired and checked by reading, not compiled, not run on hardware
//
#include <hip/hip_runtime.h>

// ---------------------------------------------------------------------------
// CLIP MLP + contrastive loss on gfx1250 (MI455X), wave32 + WMMA bf16 + TDM.
//
// Math reduction (exact): logit_scale and both row-normalizations cancel,
// so:   R = (relu(img@W1+b1) @ W2 + b2) @ txt          [4096 x 32000]
//       loss = mean_row( logsumexp(R*s) - R[tgt]*s ),  s = 1/(||R_row|| * t)
//       acc  = sum_row( argmax(R_row) == tgt )
//
// GEMMs use bf16x3 emulated-fp32: x = hi + lo (both bf16),
// acc += Ahi*Bhi + Ahi*Blo + Alo*Bhi  via v_wmma_f32_16x16x32_bf16.
// B tiles are staged into LDS by the Tensor Data Mover (double-buffered),
// shared by all 8 waves of the workgroup (8x cache-traffic reduction).
// ---------------------------------------------------------------------------

typedef __bf16 bf16_t;
typedef __attribute__((ext_vector_type(16))) __bf16 v16bf;
typedef __attribute__((ext_vector_type(8)))  __bf16 v8bf;
typedef __attribute__((ext_vector_type(8)))  float  v8f;
typedef __attribute__((ext_vector_type(4)))  unsigned int v4u;
typedef __attribute__((ext_vector_type(8)))  int    v8i;
typedef __attribute__((ext_vector_type(4)))  int    v4i;

#define B_ROWS 4096
#define D_IN   512
#define D_HID  1024
#define D_OUT  512
#define N_CLS  32000

static __device__ __forceinline__ bf16_t f2bf(float x) { return (bf16_t)x; }

static __device__ __forceinline__ v16bf join8(v8bf a, v8bf b) {
  return __builtin_shufflevector(a, b, 0,1,2,3,4,5,6,7,8,9,10,11,12,13,14,15);
}

// --------- TDM: DMA one [rows=64] x [cols=32] bf16 tile into LDS ------------
// 2D tensor descriptor per CDNA5 ISA ch.8: data_size=1 (2 bytes),
// tile_dim0=32 (K), tile_dim1=64 (N-rows), tensor_dim0_stride=K.
// Groups 2/3 zero (tile_dim2/3/4 = 0 -> unused).
static __device__ __forceinline__ void tdm_load_tile(const bf16_t* gsrc,
                                                     unsigned int lds_off,
                                                     int K, int Nrows) {
  unsigned long long ga = (unsigned long long)(size_t)gsrc;
  v4u g0 = { 1u,                                   // count=1, no gather
             lds_off,                              // bits[63:32]  lds_addr
             (unsigned int)ga,                     // bits[95:64]  ga[31:0]
             (unsigned int)(ga >> 32) | (2u << 30) // ga[56:32] | type=2
  };
  v8i g1 = { (int)(1u << 16),                                  // data_size=2B
             (int)((K & 0xFFFF) << 16),                        // tensor_dim0 lo
             (int)(((unsigned)K >> 16) | ((Nrows & 0xFFFF) << 16)), // d0 hi|d1 lo
             (int)((((unsigned)Nrows >> 16) & 0xFFFF) | (32u << 16)), // d1 hi|tile0
             64,                                               // tile_dim1
             K,                                                // dim0_stride lo
             0, 0 };
  v4i z4 = {0, 0, 0, 0};
#if defined(__clang_major__) && __clang_major__ >= 23
  v8i z8 = {0, 0, 0, 0, 0, 0, 0, 0};
  __builtin_amdgcn_tensor_load_to_lds(g0, g1, z4, z4, z8, 0);
#else
  __builtin_amdgcn_tensor_load_to_lds(g0, g1, z4, z4, 0);
#endif
}

static __device__ __forceinline__ unsigned int lds_off32(const void* p) {
  return (unsigned int)(size_t)p;   // low 32 bits of generic addr = LDS offset
}

// ---------------- elementwise fp32 -> (bf16 hi, bf16 lo) --------------------
__global__ __launch_bounds__(256) void k_split(const float* __restrict__ src,
                                               bf16_t* __restrict__ hi,
                                               bf16_t* __restrict__ lo, int n) {
  int i = blockIdx.x * blockDim.x + threadIdx.x;
  if (i < n) {
    float x = src[i];
    bf16_t h = f2bf(x);
    hi[i] = h;
    lo[i] = f2bf(x - (float)h);
  }
}

// ------------- transpose + split: src[K][N] f32 -> dstT[N][K] bf16 ----------
__global__ __launch_bounds__(256) void k_split_T(const float* __restrict__ src,
                                                 bf16_t* __restrict__ hiT,
                                                 bf16_t* __restrict__ loT,
                                                 int K, int N) {
  __shared__ float tile[32][33];
  const int tx = threadIdx.x;            // 0..31
  const int ty = threadIdx.y;            // 0..7
  const int n0 = blockIdx.x * 32;
  const int k0 = blockIdx.y * 32;
#pragma unroll
  for (int j = 0; j < 32; j += 8)
    tile[ty + j][tx] = src[(size_t)(k0 + ty + j) * N + (n0 + tx)];
  __syncthreads();
#pragma unroll
  for (int j = 0; j < 32; j += 8) {
    float x = tile[tx][ty + j];
    int n = n0 + ty + j, k = k0 + tx;
    bf16_t h = f2bf(x);
    hiT[(size_t)n * K + k] = h;
    loT[(size_t)n * K + k] = f2bf(x - (float)h);
  }
}

// ---------------- WMMA GEMM: C[M,N] = A[M,K] @ B[K,N] (B given N-major) -----
// MODE 0: C = relu(A@B + bias) -> bf16 hi/lo          (GEMM1)
// MODE 1: C = A@B + bias       -> bf16 hi/lo          (GEMM2)
// MODE 2: C = A@B              -> f32                 (GEMM3)
// Workgroup: 256 thr = 8 waves; WG tile 128(M) x 64(N); wave tile 16 x 64.
// B tile [64 x 32] bf16 (hi & lo) staged in LDS via TDM, double-buffered.
template <int MODE>
__global__ __launch_bounds__(256) void k_gemm(
    const bf16_t* __restrict__ Ahi, const bf16_t* __restrict__ Alo,
    const bf16_t* __restrict__ BThi, const bf16_t* __restrict__ BTlo,
    const float* __restrict__ bias,
    bf16_t* __restrict__ Chi, bf16_t* __restrict__ Clo,
    float* __restrict__ Cf, int M, int N, int K) {
  __shared__ __align__(16) bf16_t sBhi[2][64 * 32];   // 4 KB each
  __shared__ __align__(16) bf16_t sBlo[2][64 * 32];

  const int lane = threadIdx.x & 31;
  const int wave = threadIdx.x >> 5;
  const int row0 = blockIdx.y * 128 + wave * 16;
  const int col0 = blockIdx.x * 64;
  const int m    = lane & 15;
  const int h16  = lane >> 4;          // 0 or 1 (lane half)

  const v8f zero8 = {0.f, 0.f, 0.f, 0.f, 0.f, 0.f, 0.f, 0.f};
  v8f acc[4] = {zero8, zero8, zero8, zero8};

  const size_t arow  = (size_t)(row0 + m) * K;
  const size_t btile = (size_t)col0 * K;
  const int    ksteps = K / 32;

  // Prologue: DMA first B tile (hi+lo) into buffer 0.
  if (wave == 0) {
    tdm_load_tile(BThi + btile, lds_off32(&sBhi[0][0]), K, N);
    tdm_load_tile(BTlo + btile, lds_off32(&sBlo[0][0]), K, N);
  }

  for (int i = 0; i < ksteps; ++i) {
    const int k0  = i * 32;
    const int buf = i & 1;
    if (wave == 0) {
      if (i + 1 < ksteps) {
        const size_t off = btile + (size_t)(i + 1) * 32;
        tdm_load_tile(BThi + off, lds_off32(&sBhi[buf ^ 1][0]), K, N);
        tdm_load_tile(BTlo + off, lds_off32(&sBlo[buf ^ 1][0]), K, N);
        __builtin_amdgcn_s_wait_tensorcnt((short)2);  // current tile resident
      } else {
        __builtin_amdgcn_s_wait_tensorcnt((short)0);
      }
    }
    __syncthreads();   // publish buf to all waves

    // A fragment: ISA 16-bit A 16x32 layout.
    // lane<16 : K in {k0..k0+7, k0+16..k0+23}; lane>=16: +8 shifted halves.
    const int ca = k0 + h16 * 8;
    if (MODE == 2 && i + 1 < ksteps) __builtin_prefetch(Ahi + arow + ca + 32, 0, 0);
    v16bf ah = join8(*(const v8bf*)(Ahi + arow + ca),
                     *(const v8bf*)(Ahi + arow + ca + 16));
    v16bf al = join8(*(const v8bf*)(Alo + arow + ca),
                     *(const v8bf*)(Alo + arow + ca + 16));
#pragma unroll
    for (int t = 0; t < 4; ++t) {
      // B frag: same K order as A (lane<16: k0..k0+7, k0+16..k0+23; lane>=16: +8).
      const int boff = (t * 16 + m) * 32 + h16 * 8;
      v16bf bh = join8(*(const v8bf*)&sBhi[buf][boff],
                       *(const v8bf*)&sBhi[buf][boff + 16]);
      v16bf bl = join8(*(const v8bf*)&sBlo[buf][boff],
                       *(const v8bf*)&sBlo[buf][boff + 16]);
      acc[t] = __builtin_amdgcn_wmma_f32_16x16x32_bf16(
          false, ah, false, bh, (short)0, acc[t], false, false);
      acc[t] = __builtin_amdgcn_wmma_f32_16x16x32_bf16(
          false, ah, false, bl, (short)0, acc[t], false, false);
      acc[t] = __builtin_amdgcn_wmma_f32_16x16x32_bf16(
          false, al, false, bh, (short)0, acc[t], false, false);
    }
    __syncthreads();   // all reads of buf done before it is overwritten
  }

  // Epilogue. C/D layout: VGPR r, lane<16 -> (M=r, N=lane); lane>=16 -> (M=8+r).
#pragma unroll
  for (int t = 0; t < 4; ++t) {
    const int col = col0 + t * 16 + m;
    float bv = (MODE == 2) ? 0.f : bias[col];
#pragma unroll
    for (int r = 0; r < 8; ++r) {
      const int row = row0 + r + h16 * 8;
      float v = acc[t][r];
      if (MODE != 2) v += bv;
      if (MODE == 0) v = v > 0.f ? v : 0.f;
      if (MODE == 2) {
        Cf[(size_t)row * N + col] = v;
      } else {
        bf16_t h = f2bf(v);
        Chi[(size_t)row * N + col] = h;
        Clo[(size_t)row * N + col] = f2bf(v - (float)h);
      }
    }
  }
}

// ------------- per-row: ||R||, max, argmax, stable logsumexp ----------------
__global__ __launch_bounds__(256) void k_rowstats(const float* __restrict__ R,
                                                  const int* __restrict__ tgt,
                                                  const int* __restrict__ tptr,
                                                  float* __restrict__ loss_row,
                                                  float* __restrict__ acc_row) {
  const int row = blockIdx.x;
  const int tid = threadIdx.x;
  const float* r = R + (size_t)row * N_CLS;
  __shared__ float s_ss[256];
  __shared__ float s_mx[256];
  __shared__ int   s_am[256];
  __shared__ float s_se[256];

  float ss = 0.f, mx = -__builtin_inff();
  int am = N_CLS;
  for (int i = tid; i < N_CLS; i += 256) {
    float x = r[i];
    ss += x * x;
    if (x > mx) { mx = x; am = i; }
  }
  s_ss[tid] = ss; s_mx[tid] = mx; s_am[tid] = am;
  __syncthreads();
  for (int s = 128; s > 0; s >>= 1) {
    if (tid < s) {
      s_ss[tid] += s_ss[tid + s];
      float m2 = s_mx[tid + s]; int a2 = s_am[tid + s];
      if (m2 > s_mx[tid] || (m2 == s_mx[tid] && a2 < s_am[tid])) {
        s_mx[tid] = m2; s_am[tid] = a2;
      }
    }
    __syncthreads();
  }
  const float nrm   = sqrtf(s_ss[0]);
  const float tval  = (float)(*tptr);
  const float scale = 1.f / (nrm * tval);
  const float mxv   = s_mx[0];

  float se = 0.f;
  for (int i = tid; i < N_CLS; i += 256) se += expf((r[i] - mxv) * scale);
  s_se[tid] = se;
  __syncthreads();
  for (int s = 128; s > 0; s >>= 1) {
    if (tid < s) s_se[tid] += s_se[tid + s];
    __syncthreads();
  }
  if (tid == 0) {
    const int tg = tgt[row];
    const float lse = mxv * scale + logf(s_se[0]);
    loss_row[row] = lse - r[tg] * scale;
    acc_row[row]  = (s_am[0] == tg) ? 1.f : 0.f;
  }
}

// ---------------- final deterministic reduce -> d_out[2] --------------------
__global__ __launch_bounds__(256) void k_final(const float* __restrict__ loss_row,
                                               const float* __restrict__ acc_row,
                                               float* __restrict__ out) {
  __shared__ float sl[256], sa[256];
  const int tid = threadIdx.x;
  float l = 0.f, a = 0.f;
  for (int i = tid; i < B_ROWS; i += 256) { l += loss_row[i]; a += acc_row[i]; }
  sl[tid] = l; sa[tid] = a;
  __syncthreads();
  for (int s = 128; s > 0; s >>= 1) {
    if (tid < s) { sl[tid] += sl[tid + s]; sa[tid] += sa[tid + s]; }
    __syncthreads();
  }
  if (tid == 0) { out[0] = sl[0] / (float)B_ROWS; out[1] = sa[0]; }
}

// ---------------------------------------------------------------------------
extern "C" void kernel_launch(void* const* d_in, const int* in_sizes, int n_in,
                              void* d_out, int out_size, void* d_ws, size_t ws_size,
                              hipStream_t stream) {
  const float* img  = (const float*)d_in[0];   // [4096, 512]
  const float* txt  = (const float*)d_in[1];   // [512, 32000]
  const int*   tgt  = (const int*)d_in[2];     // [4096]
  const float* W1   = (const float*)d_in[3];   // [512, 1024]
  const float* b1   = (const float*)d_in[4];   // [1024]
  const float* W2   = (const float*)d_in[5];   // [1024, 512]
  const float* b2   = (const float*)d_in[6];   // [512]
  // d_in[7] = logit_scale: cancels analytically (see header comment)
  const int*   tptr = (const int*)d_in[8];     // temperature t
  float* out = (float*)d_out;

  char*  ws  = (char*)d_ws;
  size_t off = 0;
  auto alloc = [&](size_t bytes) -> char* {
    char* p = ws + off;
    off += (bytes + 255) & ~(size_t)255;
    return p;
  };

  float*  R        = (float*)alloc((size_t)B_ROWS * N_CLS * 4);     // 524 MB
  bf16_t* img_hi   = (bf16_t*)alloc((size_t)B_ROWS * D_IN * 2);
  bf16_t* img_lo   = (bf16_t*)alloc((size_t)B_ROWS * D_IN * 2);
  bf16_t* W1T_hi   = (bf16_t*)alloc((size_t)D_HID * D_IN * 2);
  bf16_t* W1T_lo   = (bf16_t*)alloc((size_t)D_HID * D_IN * 2);
  bf16_t* W2T_hi   = (bf16_t*)alloc((size_t)D_OUT * D_HID * 2);
  bf16_t* W2T_lo   = (bf16_t*)alloc((size_t)D_OUT * D_HID * 2);
  bf16_t* txtT_hi  = (bf16_t*)alloc((size_t)N_CLS * D_OUT * 2);     // 32 MB
  bf16_t* txtT_lo  = (bf16_t*)alloc((size_t)N_CLS * D_OUT * 2);     // 32 MB
  bf16_t* h_hi     = (bf16_t*)alloc((size_t)B_ROWS * D_HID * 2);
  bf16_t* h_lo     = (bf16_t*)alloc((size_t)B_ROWS * D_HID * 2);
  bf16_t* mlp_hi   = (bf16_t*)alloc((size_t)B_ROWS * D_OUT * 2);
  bf16_t* mlp_lo   = (bf16_t*)alloc((size_t)B_ROWS * D_OUT * 2);
  float*  loss_row = (float*)alloc((size_t)B_ROWS * 4);
  float*  acc_row  = (float*)alloc((size_t)B_ROWS * 4);

  // 1) precision split / transposed-split of operands
  k_split<<<(B_ROWS * D_IN + 255) / 256, 256, 0, stream>>>(img, img_hi, img_lo,
                                                           B_ROWS * D_IN);
  k_split_T<<<dim3(D_HID / 32, D_IN / 32), dim3(32, 8), 0, stream>>>(
      W1, W1T_hi, W1T_lo, D_IN, D_HID);
  k_split_T<<<dim3(D_OUT / 32, D_HID / 32), dim3(32, 8), 0, stream>>>(
      W2, W2T_hi, W2T_lo, D_HID, D_OUT);
  k_split_T<<<dim3(N_CLS / 32, D_OUT / 32), dim3(32, 8), 0, stream>>>(
      txt, txtT_hi, txtT_lo, D_OUT, N_CLS);

  // 2) h = relu(img @ W1 + b1)
  k_gemm<0><<<dim3(D_HID / 64, B_ROWS / 128), 256, 0, stream>>>(
      img_hi, img_lo, W1T_hi, W1T_lo, b1, h_hi, h_lo, nullptr,
      B_ROWS, D_HID, D_IN);
  // 3) mlp = h @ W2 + b2
  k_gemm<1><<<dim3(D_OUT / 64, B_ROWS / 128), 256, 0, stream>>>(
      h_hi, h_lo, W2T_hi, W2T_lo, b2, mlp_hi, mlp_lo, nullptr,
      B_ROWS, D_OUT, D_HID);
  // 4) R = mlp @ txt   (the 134-GFLOP GEMM)
  k_gemm<2><<<dim3(N_CLS / 64, B_ROWS / 128), 256, 0, stream>>>(
      mlp_hi, mlp_lo, txtT_hi, txtT_lo, nullptr, nullptr, nullptr, R,
      B_ROWS, N_CLS, D_OUT);

  // 5) per-row logsumexp / argmax, 6) final reduce
  k_rowstats<<<B_ROWS, 256, 0, stream>>>(R, tgt, tptr, loss_row, acc_row);
  k_final<<<1, 256, 0, stream>>>(loss_row, acc_row, out);
}
